// dual_att_layer_68753836474511
// MI455X (gfx1250) — compile-verified, matrix operand order checked by reading, not run on hardware
//
#include <hip/hip_runtime.h>
#include <hip/hip_bf16.h>
#include <stdint.h>

typedef __attribute__((ext_vector_type(16))) _Float16 v16h;
typedef __attribute__((ext_vector_type(8)))  float    v8f;

#define N_NODES 8192
#define DDIM    256
#define HDIM    128
#define NEG_BIG (-1.0e9f)
#define SLOPE   0.01f
#define FLT_BIG 3.402823466e+38f
#define CHUNK   128                    // K columns per main-loop iteration
#define NITER   (N_NODES / CHUNK)      // 64
#define MROWS   32                     // output rows per block
#define LDS_STRIDE 68                  // dwords per coef row (64 + pad)

__device__ __forceinline__ float leaky(float x) { return x > 0.f ? x : x * SLOPE; }

// ---------------------------------------------------------------------------
// Kernel A: fold linear layer + attention vectors:
//   u1[d] = sum_k W[k,d]*w1[k]; c1 = b.w1 + b1  (same for 2)
// ---------------------------------------------------------------------------
__global__ void prep_uv(const float* __restrict__ W, const float* __restrict__ b,
                        const float* __restrict__ w1, const float* __restrict__ b1,
                        const float* __restrict__ w2, const float* __restrict__ b2,
                        float* __restrict__ uv) {
  int d = threadIdx.x;  // 256 threads
  float a1 = 0.f, a2 = 0.f;
  for (int k = 0; k < HDIM; ++k) {
    float w = W[k * DDIM + d];
    a1 += w * w1[k];
    a2 += w * w2[k];
  }
  uv[d] = a1;
  uv[DDIM + d] = a2;
  if (d == 0) {
    float c = 0.f;
    for (int k = 0; k < HDIM; ++k) c += b[k] * w1[k];
    uv[2 * DDIM] = c + b1[0];
  }
  if (d == 1) {
    float c = 0.f;
    for (int k = 0; k < HDIM; ++k) c += b[k] * w2[k];
    uv[2 * DDIM + 1] = c + b2[0];
  }
}

// ---------------------------------------------------------------------------
// Kernel B: f1[i] = inlayer2[i,:].u1 + c1 ; f2[i] = inlayer2[i,:].u2 + c2
// one wave32 per row, shuffle reduction
// ---------------------------------------------------------------------------
__global__ void node_feats(const float* __restrict__ inlayer2, const float* __restrict__ uv,
                           float* __restrict__ f1, float* __restrict__ f2) {
  int wave = threadIdx.x >> 5, lane = threadIdx.x & 31;
  int row = blockIdx.x * 8 + wave;
  const float* x = inlayer2 + (size_t)row * DDIM;
  float a1 = 0.f, a2 = 0.f;
  for (int d = lane; d < DDIM; d += 32) {
    float xv = x[d];
    a1 += xv * uv[d];
    a2 += xv * uv[DDIM + d];
  }
  for (int off = 16; off; off >>= 1) {
    a1 += __shfl_xor(a1, off, 32);
    a2 += __shfl_xor(a2, off, 32);
  }
  if (lane == 0) {
    f1[row] = a1 + uv[2 * DDIM];
    f2[row] = a2 + uv[2 * DDIM + 1];
  }
}

// ---------------------------------------------------------------------------
// Kernel C: per-row online softmax stats + adjacency bit-compression.
//   s_ij = leaky(f1[i]+f2[j]) + (adj ? 0 : -1e9); store m_i, 1/Z_i and a
//   1-bit/col bitmap (via wave32 ballot) so the main pass never touches the
//   268 MB adj matrix again (bitmap = 8 MB, L2-resident).
// adj is streamed with non-temporal loads to avoid thrashing the 192 MB L2.
// ---------------------------------------------------------------------------
__global__ void row_stats(const int* __restrict__ adj, const float* __restrict__ f1,
                          const float* __restrict__ f2, float* __restrict__ mrow,
                          float* __restrict__ invz, uint32_t* __restrict__ bitmap) {
  __shared__ float sm[256], ss[256];
  int i = blockIdx.x, tid = threadIdx.x;
  float f1i = f1[i];
  const int* arow = adj + (size_t)i * N_NODES;
  uint32_t* bmrow = bitmap + (size_t)i * (N_NODES / 32);
  float mt = -FLT_BIG, st = 0.f;
  for (int j = tid; j < N_NODES; j += 256) {
    int av = __builtin_nontemporal_load(&arow[j]);
    uint32_t bits = (uint32_t)__ballot(av > 0);
    if ((tid & 31) == 0) bmrow[j >> 5] = bits;
    float s = leaky(f1i + f2[j]) + (av > 0 ? 0.f : NEG_BIG);
    // branchless online-softmax update (single exp)
    float M = fmaxf(mt, s);
    float d = __expf(fminf(mt, s) - M);
    float ca = (s > mt) ? d : 1.f;
    float cb = (s > mt) ? 1.f : d;
    st = st * ca + cb;
    mt = M;
  }
  sm[tid] = mt; ss[tid] = st;
  __syncthreads();
  for (int off = 128; off > 0; off >>= 1) {
    if (tid < off) {
      float m1 = sm[tid], s1 = ss[tid];
      float m2 = sm[tid + off], s2 = ss[tid + off];
      float M = fmaxf(m1, m2);
      sm[tid] = M;
      ss[tid] = s1 * __expf(m1 - M) + s2 * __expf(m2 - M);
    }
    __syncthreads();
  }
  if (tid == 0) { mrow[i] = sm[0]; invz[i] = 1.f / ss[0]; }
}

// ---------------------------------------------------------------------------
// Kernel D: pre-swizzle inlayer (f32 -> f16) into WMMA B-fragment layout.
// 16-bit operand layout (ISA 7.12.2): VGPR v holds K = 16*(v>>2) + 2*(v&3)
// + 8*(lane>=16), two K values per dword. Stored as
//   packedB[(((kb*16+nt)*32 + lane)*8 + v]   (dwords, kb = 32-col K block)
// so the main loop does two lane-contiguous b128 loads per fragment.
// ---------------------------------------------------------------------------
__global__ void pack_b(const float* __restrict__ inlayer, uint32_t* __restrict__ packedB) {
  uint32_t t = blockIdx.x * 256 + threadIdx.x;  // 1M words total
  uint32_t v = t & 7, lane = (t >> 3) & 31, nt = (t >> 8) & 15, kb = t >> 12;
  uint32_t n = lane & 15, hi = lane >> 4;
  uint32_t k = ((v >> 2) << 4) | ((v & 3) << 1) | (hi << 3);
  uint32_t j0 = kb * 32 + k, d = nt * 16 + n;
  float v0 = inlayer[(size_t)j0 * DDIM + d];
  float v1 = inlayer[(size_t)(j0 + 1) * DDIM + d];
  union { _Float16 h[2]; uint32_t u; } p;
  p.h[0] = (_Float16)v0;
  p.h[1] = (_Float16)v1;
  packedB[t] = p.u;
}

// ---------------------------------------------------------------------------
// Kernel E: main fused pass.  Block = 32 output rows x full D=256.
// 8 waves, each owns 2 M-tiles x 2 N-tiles (4 x v8f accumulators); B
// fragments are reused across M-tiles to halve L2 traffic.
// Per 128-column K-chunk: block computes the 32x128 coef tile
//   coef = bit ? exp(leaky(f1+f2) - m_i) : 0   (f16, padded LDS, branchless)
// into a double-buffered LDS tile (one barrier per chunk), prefetching the
// next chunk's bitmap/f2 before the barrier so global loads overlap the
// 16 v_wmma_f32_16x16x32_f16 per wave per chunk.  Writeout scales by 1/Z.
// ---------------------------------------------------------------------------
__global__ void __launch_bounds__(256)
gat_main(const uint32_t* __restrict__ bitmap, const float* __restrict__ f1,
         const float* __restrict__ f2, const float* __restrict__ mrow,
         const float* __restrict__ invz, const uint32_t* __restrict__ packedB,
         float* __restrict__ out) {
  __shared__ uint32_t ldsA[2][MROWS * LDS_STRIDE];
  const int i0 = blockIdx.x * MROWS;
  const int tid = threadIdx.x;
  const int lane = tid & 31, wave = tid >> 5;
  const int hi = lane >> 4, mm = lane & 15;
  const int nt0 = wave * 2, nt1 = nt0 + 1;

  // producer mapping: thread -> (row wm, 16 cols starting at c0)
  const int wm = tid >> 3;            // 0..31
  const int tc = tid & 7;             // 16-col group within chunk
  const float f1i = f1[i0 + wm];
  const float mi = mrow[i0 + wm];
  const unsigned short* bmRow =
      reinterpret_cast<const unsigned short*>(bitmap + (size_t)(i0 + wm) * (N_NODES / 32));
  const float4* f2v = reinterpret_cast<const float4*>(f2);

  v8f acc00 = {}, acc01 = {}, acc10 = {}, acc11 = {};
  union ABu { uint32_t u[8]; v16h h; };
  union P16 { _Float16 h[16]; uint4 q[2]; };

  // prefetch chunk 0
  unsigned int bm = bmRow[tc];
  float4 F0 = f2v[tc * 4 + 0], F1 = f2v[tc * 4 + 1];
  float4 F2 = f2v[tc * 4 + 2], F3 = f2v[tc * 4 + 3];

  int p = 0;
  for (int jb = 0; jb < NITER; ++jb) {
    // ---- producer: 16 coefs (exp computed unconditionally, mask via select)
    {
      float fr[16] = {F0.x, F0.y, F0.z, F0.w, F1.x, F1.y, F1.z, F1.w,
                      F2.x, F2.y, F2.z, F2.w, F3.x, F3.y, F3.z, F3.w};
      P16 pk;
#pragma unroll
      for (int l = 0; l < 16; ++l) {
        float ex = __expf(leaky(f1i + fr[l]) - mi);
        float e = ((bm >> l) & 1u) ? ex : 0.f;
        pk.h[l] = (_Float16)e;
      }
      uint4* dst = reinterpret_cast<uint4*>(&ldsA[p][wm * LDS_STRIDE + tc * 8]);
      dst[0] = pk.q[0];
      dst[1] = pk.q[1];
    }
    // ---- prefetch next chunk (in flight during the WMMA phase)
    if (jb + 1 < NITER) {
      int o = (jb + 1) * 32 + tc * 4;
      bm = bmRow[(jb + 1) * 8 + tc];
      F0 = f2v[o + 0]; F1 = f2v[o + 1]; F2 = f2v[o + 2]; F3 = f2v[o + 3];
    }
    __syncthreads();

    // ---- consumer: 4 K-subtiles of 32; A frag = two ds_load_b128
    const uint32_t* buf = ldsA[p];
#pragma unroll
    for (int s4 = 0; s4 < 4; ++s4) {
      ABu a0, a1;
      int base0 = mm * LDS_STRIDE + s4 * 16 + hi * 4;
      int base1 = (16 + mm) * LDS_STRIDE + s4 * 16 + hi * 4;
      *reinterpret_cast<uint4*>(&a0.u[0]) = *reinterpret_cast<const uint4*>(&buf[base0]);
      *reinterpret_cast<uint4*>(&a0.u[4]) = *reinterpret_cast<const uint4*>(&buf[base0 + 8]);
      *reinterpret_cast<uint4*>(&a1.u[0]) = *reinterpret_cast<const uint4*>(&buf[base1]);
      *reinterpret_cast<uint4*>(&a1.u[4]) = *reinterpret_cast<const uint4*>(&buf[base1 + 8]);

      int kb = jb * 4 + s4;
      ABu b0, b1;
      const uint4* bp0 = reinterpret_cast<const uint4*>(
          packedB + (((size_t)(kb * 16 + nt0) * 32 + lane) * 8));
      const uint4* bp1 = reinterpret_cast<const uint4*>(
          packedB + (((size_t)(kb * 16 + nt1) * 32 + lane) * 8));
      *reinterpret_cast<uint4*>(&b0.u[0]) = bp0[0];
      *reinterpret_cast<uint4*>(&b0.u[4]) = bp0[1];
      *reinterpret_cast<uint4*>(&b1.u[0]) = bp1[0];
      *reinterpret_cast<uint4*>(&b1.u[4]) = bp1[1];

      acc00 = __builtin_amdgcn_wmma_f32_16x16x32_f16(false, a0.h, false, b0.h,
                                                     (short)0, acc00, false, false);
      acc01 = __builtin_amdgcn_wmma_f32_16x16x32_f16(false, a0.h, false, b1.h,
                                                     (short)0, acc01, false, false);
      acc10 = __builtin_amdgcn_wmma_f32_16x16x32_f16(false, a1.h, false, b0.h,
                                                     (short)0, acc10, false, false);
      acc11 = __builtin_amdgcn_wmma_f32_16x16x32_f16(false, a1.h, false, b1.h,
                                                     (short)0, acc11, false, false);
    }
    p ^= 1;
  }

  // C layout: VGPR r -> M = r + 8*hi, N = lane&15. Scale by 1/Z and store.
#pragma unroll
  for (int r = 0; r < 8; ++r) {
    int row0 = i0 + r + hi * 8;
    int row1 = row0 + 16;
    float s0 = invz[row0], s1 = invz[row1];
    out[(size_t)row0 * DDIM + nt0 * 16 + mm] = acc00[r] * s0;
    out[(size_t)row0 * DDIM + nt1 * 16 + mm] = acc01[r] * s0;
    out[(size_t)row1 * DDIM + nt0 * 16 + mm] = acc10[r] * s1;
    out[(size_t)row1 * DDIM + nt1 * 16 + mm] = acc11[r] * s1;
  }
}

// ---------------------------------------------------------------------------
extern "C" void kernel_launch(void* const* d_in, const int* in_sizes, int n_in,
                              void* d_out, int out_size, void* d_ws, size_t ws_size,
                              hipStream_t stream) {
  const float* inlayer  = (const float*)d_in[0];
  const float* inlayer2 = (const float*)d_in[1];
  const int*   adj      = (const int*)d_in[2];
  const float* W  = (const float*)d_in[3];
  const float* b  = (const float*)d_in[4];
  const float* w1 = (const float*)d_in[5];
  const float* b1 = (const float*)d_in[6];
  const float* w2 = (const float*)d_in[7];
  const float* b2 = (const float*)d_in[8];
  float* out = (float*)d_out;

  // workspace (floats): [uv:1024][f1:8192][f2:8192][m:8192][invz:8192]
  //                     [packedB: 1M dwords = 4MB][bitmap: 2M dwords = 8MB]
  float* ws_f = (float*)d_ws;
  float* uv   = ws_f;
  float* f1   = ws_f + 1024;
  float* f2   = f1 + N_NODES;
  float* mr   = f2 + N_NODES;
  float* iz   = mr + N_NODES;
  uint32_t* packedB = (uint32_t*)(iz + N_NODES);
  uint32_t* bitmap  = packedB + (N_NODES / 32) * 16 * 32 * 8;  // 1M dwords later

  prep_uv   <<<1, 256, 0, stream>>>(W, b, w1, b1, w2, b2, uv);
  node_feats<<<N_NODES / 8, 256, 0, stream>>>(inlayer2, uv, f1, f2);
  pack_b    <<<(N_NODES / 32) * 16 * 32 * 8 / 256, 256, 0, stream>>>(inlayer, packedB);
  row_stats <<<N_NODES, 256, 0, stream>>>(adj, f1, f2, mr, iz, bitmap);
  gat_main  <<<N_NODES / MROWS, 256, 0, stream>>>(bitmap, f1, f2, mr, iz, packedB, out);
}
